// FastBloomModel_87419764343115
// MI455X (gfx1250) — matrix operand order repaired and checked by reading, not compiled, not run on hardware
//
#include <hip/hip_runtime.h>
#include <hip/hip_bf16.h>
#include <math.h>

// ---------------------------------------------------------------------------
// FastBloom forward for MI455X (gfx1250, wave32, WMMA).
// GEMMs: v_wmma_f32_16x16x32_bf16, bf16 inputs / f32 accumulate.
// Block tile 128x128: B fragments staged through LDS (double buffered) so B
// weight data is read from L2 once per 128 output rows instead of once per 16.
// ---------------------------------------------------------------------------

typedef __attribute__((ext_vector_type(16))) __bf16 v16bf;
typedef __attribute__((ext_vector_type(8)))  float  v8f;

#define BATCH   2
#define SEQ     1024
#define TOKENS  (BATCH * SEQ)     // 2048
#define DMODEL  1024
#define NHEAD   16
#define HDIM    64
#define QKVD    (3 * DMODEL)      // 3072
#define FFD     (4 * DMODEL)      // 4096
#define NLAYER  4

struct B32B { uint4 a, b; };      // 32 bytes == one v16bf fragment slice

__device__ __forceinline__ float bloom_gelu(float x) {
  return x * 0.5f * (1.0f + tanhf(0.79788456f * x * (1.0f + 0.044715f * x * x)));
}

// ---------------------------------------------------------------------------
// Repack a K x N row-major f32 weight matrix into bf16 WMMA B-fragment layout.
// Layout: [nb = n/16][kb = k/32][lane 0..31][16 bf16].
// B-matrix (this code's contiguous K order, matching A): lanes 0-15 hold column n, K = kb*32 + 0..15;
// lanes 16-31 hold column n, K = kb*32 + 16..31 (16 contiguous K per lane).
// ---------------------------------------------------------------------------
__global__ __launch_bounds__(256) void pack_weight_kernel(
    const float* __restrict__ W, __bf16* __restrict__ out, int K, int N)
{
  int idx = blockIdx.x * 256 + threadIdx.x;
  int Kb = K >> 5;
  int total = (N >> 4) * Kb * 32;
  if (idx >= total) return;
  int lane = idx & 31;
  int kb = (idx >> 5) % Kb;
  int nb = (idx >> 5) / Kb;
  int n  = (nb << 4) + (lane & 15);
  int k0 = (kb << 5) + ((lane >> 4) << 4);
  v16bf v;
  #pragma unroll
  for (int e = 0; e < 16; ++e)
    v[e] = (__bf16)W[(size_t)(k0 + e) * N + n];
  ((v16bf*)out)[idx] = v;
}

// ---------------------------------------------------------------------------
// WMMA GEMM: C[M,N] = A[M,K] (f32 -> bf16 on the fly) x Wp (packed bf16)
//            + bias, epilogue (op: 0 = bias, 1 = +gelu, 2 = +residual).
// Block = 8 waves, tile = 128(M) x 128(N). Per K-step (32):
//   - 256 threads cooperatively stage 8 B fragments (8 KB) into LDS,
//     double-buffered: global->regs for step kb+1 issued before compute of kb.
//   - each wave builds one A fragment from 4x float4 loads (contiguous
//     16-bit A layout) and runs 8 WMMAs against the shared B fragments.
// ---------------------------------------------------------------------------
__global__ __launch_bounds__(256) void wmma_gemm_kernel(
    const float* __restrict__ A, const __bf16* __restrict__ Wp,
    const float* __restrict__ bias, const float* __restrict__ resid,
    float* __restrict__ C, int M, int K, int N, int op)
{
  __shared__ uint4 smem[1024];             // 16 KB = 2 x (8 frags x 32 lanes x 32B)

  const int tid  = threadIdx.x;
  const int lane = tid & 31;
  const int wave = tid >> 5;
  const int nblk = N >> 7;                 // 128-wide N blocks
  const int bm   = blockIdx.x / nblk;
  const int bn   = blockIdx.x - bm * nblk;

  const int Kb = K >> 5;

  // ---- A fragment addressing (per wave: rows bm*128 + wave*16 .. +15) ----
  const int mrowA  = bm * 128 + wave * 16 + (lane & 15);
  const int khalfA = (lane >> 4) << 4;     // 0 or 16
  const float* Arow = A + (size_t)mrowA * K + khalfA;

  // ---- cooperative B loader role: thread t stages fragment f=t>>5, lane t&31
  const int lf    = tid >> 5;              // fragment 0..7
  const int nbBase = bn * 8;               // first 16-col fragment of this block
  const B32B* Bg  = (const B32B*)Wp;
  const size_t bgBase = ((size_t)(nbBase + lf) * Kb) * 32 + (tid & 31);

  // prologue: stage kb = 0 into buffer 0
  B32B stage = Bg[bgBase];
  smem[tid * 2 + 0] = stage.a;
  smem[tid * 2 + 1] = stage.b;

  v8f acc[8];
  #pragma unroll
  for (int f = 0; f < 8; ++f)
    #pragma unroll
    for (int i = 0; i < 8; ++i) acc[f][i] = 0.0f;

  int cur = 0;
  for (int kb = 0; kb < Kb; ++kb) {
    __syncthreads();                       // buffer `cur` fully written

    // issue next B stage (global -> regs) early to hide latency
    if (kb + 1 < Kb) stage = Bg[bgBase + (size_t)(kb + 1) * 32];

    // A fragment: lanes 0-15 K=0..15, lanes 16-31 K=16..31
    float4 a0 = *(const float4*)(Arow + 0);
    float4 a1 = *(const float4*)(Arow + 4);
    float4 a2 = *(const float4*)(Arow + 8);
    float4 a3 = *(const float4*)(Arow + 12);
    Arow += 32;
    v16bf af;
    af[0]  = (__bf16)a0.x; af[1]  = (__bf16)a0.y; af[2]  = (__bf16)a0.z; af[3]  = (__bf16)a0.w;
    af[4]  = (__bf16)a1.x; af[5]  = (__bf16)a1.y; af[6]  = (__bf16)a1.z; af[7]  = (__bf16)a1.w;
    af[8]  = (__bf16)a2.x; af[9]  = (__bf16)a2.y; af[10] = (__bf16)a2.z; af[11] = (__bf16)a2.w;
    af[12] = (__bf16)a3.x; af[13] = (__bf16)a3.y; af[14] = (__bf16)a3.z; af[15] = (__bf16)a3.w;

    // 8 WMMAs: A fragment reused against 8 shared B fragments from LDS
    const uint4* lb = &smem[cur * 512];
    #pragma unroll
    for (int f = 0; f < 8; f += 2) {
      B32B t0, t1;
      t0.a = lb[(f * 32 + lane) * 2 + 0];
      t0.b = lb[(f * 32 + lane) * 2 + 1];
      t1.a = lb[((f + 1) * 32 + lane) * 2 + 0];
      t1.b = lb[((f + 1) * 32 + lane) * 2 + 1];
      v16bf b0 = __builtin_bit_cast(v16bf, t0);
      v16bf b1 = __builtin_bit_cast(v16bf, t1);
      acc[f]     = __builtin_amdgcn_wmma_f32_16x16x32_bf16(false, af, false, b0, (short)0, acc[f],     false, false);
      acc[f + 1] = __builtin_amdgcn_wmma_f32_16x16x32_bf16(false, af, false, b1, (short)0, acc[f + 1], false, false);
    }

    // write staged regs into the other buffer
    if (kb + 1 < Kb) {
      int nxt = cur ^ 1;
      smem[nxt * 512 + tid * 2 + 0] = stage.a;
      smem[nxt * 512 + tid * 2 + 1] = stage.b;
    }
    cur ^= 1;
  }

  // C/D layout: VGPR r, lanes 0-15 -> (M = r, N = lane); lanes 16-31 -> M = r+8.
  const int mBase = bm * 128 + wave * 16 + ((lane >> 4) << 3);
  const int nlane = lane & 15;
  #pragma unroll
  for (int f = 0; f < 8; ++f) {
    int n = (bn << 7) + (f << 4) + nlane;
    float bv = bias[n];
    #pragma unroll
    for (int r = 0; r < 8; ++r) {
      int m = mBase + r;
      float v = acc[f][r] + bv;
      if (op == 1)      v = bloom_gelu(v);
      else if (op == 2) v += resid[(size_t)m * N + n];
      C[(size_t)m * N + n] = v;
    }
  }
}

// ---------------------------------------------------------------------------
// Block-wide sum reduction (blockDim.x == 256).
// ---------------------------------------------------------------------------
__device__ __forceinline__ float block_sum(float v, float* sbuf) {
  int tid = threadIdx.x;
  sbuf[tid] = v;
  __syncthreads();
  for (int s = 128; s > 0; s >>= 1) {
    if (tid < s) sbuf[tid] += sbuf[tid + s];
    __syncthreads();
  }
  float r = sbuf[0];
  __syncthreads();
  return r;
}

// LayerNorm over D=1024, one block (256 threads) per token.
__global__ __launch_bounds__(256) void layernorm_kernel(
    const float* __restrict__ x, const float* __restrict__ g,
    const float* __restrict__ b, float* __restrict__ y)
{
  __shared__ float sbuf[256];
  size_t t = blockIdx.x;
  const float* xr = x + t * DMODEL;
  float vals[4], s = 0.0f;
  #pragma unroll
  for (int i = 0; i < 4; ++i) { vals[i] = xr[threadIdx.x + 256 * i]; s += vals[i]; }
  float mean = block_sum(s, sbuf) * (1.0f / DMODEL);
  float vs = 0.0f;
  #pragma unroll
  for (int i = 0; i < 4; ++i) { float d = vals[i] - mean; vs += d * d; }
  float var = block_sum(vs, sbuf) * (1.0f / DMODEL);
  float rstd = rsqrtf(var + 1e-5f);
  #pragma unroll
  for (int i = 0; i < 4; ++i) {
    int c = threadIdx.x + 256 * i;
    y[t * DMODEL + c] = (vals[i] - mean) * rstd * g[c] + b[c];
  }
}

// Embedding gather + LayerNorm fused; one block per token.
__global__ __launch_bounds__(256) void embed_ln_kernel(
    const int* __restrict__ ids, const float* __restrict__ wte,
    const float* __restrict__ g, const float* __restrict__ b,
    float* __restrict__ h)
{
  __shared__ float sbuf[256];
  size_t t = blockIdx.x;
  const float* xr = wte + (size_t)ids[t] * DMODEL;
  float vals[4], s = 0.0f;
  #pragma unroll
  for (int i = 0; i < 4; ++i) { vals[i] = xr[threadIdx.x + 256 * i]; s += vals[i]; }
  float mean = block_sum(s, sbuf) * (1.0f / DMODEL);
  float vs = 0.0f;
  #pragma unroll
  for (int i = 0; i < 4; ++i) { float d = vals[i] - mean; vs += d * d; }
  float var = block_sum(vs, sbuf) * (1.0f / DMODEL);
  float rstd = rsqrtf(var + 1e-5f);
  #pragma unroll
  for (int i = 0; i < 4; ++i) {
    int c = threadIdx.x + 256 * i;
    h[t * DMODEL + c] = (vals[i] - mean) * rstd * g[c] + b[c];
  }
}

// ALiBi positions: pos[b,s] = mask[b,s] * (cumsum(mask)[b,s] - 1)
__global__ void positions_kernel(const float* __restrict__ m, float* __restrict__ pos)
{
  int b = blockIdx.x, s = threadIdx.x;
  float c = 0.0f;
  for (int j = 0; j <= s; ++j) c += m[b * SEQ + j];
  pos[b * SEQ + s] = m[b * SEQ + s] * (c - 1.0f);
}

// ---------------------------------------------------------------------------
// Fused attention: one block per (b, h, q). The per-layer lnum factors cancel:
//   score = q.k / sqrt(HD) + slope_h * pos_k + mask_k(+causal)
// then softmax over k, then ctx = probs . v, written as [b,s,h*64+d].
// ---------------------------------------------------------------------------
__global__ __launch_bounds__(256) void attention_kernel(
    const float* __restrict__ qkv, const float* __restrict__ amask,
    const float* __restrict__ pos, float* __restrict__ ctx)
{
  __shared__ float qs[HDIM];
  __shared__ float sc[SEQ];
  __shared__ float red[256];
  __shared__ float cpart[4][HDIM];

  const int q = blockIdx.x, h = blockIdx.y, b = blockIdx.z;
  const int tid = threadIdx.x;

  const float* qp = qkv + ((size_t)(b * SEQ + q) * QKVD) + h * (3 * HDIM);
  if (tid < HDIM) qs[tid] = qp[tid];
  __syncthreads();

  const float slope = exp2f(-0.5f * (float)(h + 1));

  for (int kk = tid; kk < SEQ; kk += 256) {
    const float* kp = qkv + ((size_t)(b * SEQ + kk) * QKVD) + h * (3 * HDIM) + HDIM;
    float d = 0.0f;
    #pragma unroll
    for (int dd = 0; dd < HDIM; ++dd) d += qs[dd] * kp[dd];
    float mterm = -10000.0f * (1.0f - amask[b * SEQ + kk]) + (kk > q ? -10000.0f : 0.0f);
    sc[kk] = d * 0.125f + slope * pos[b * SEQ + kk] + mterm;
  }
  __syncthreads();

  // max reduce
  float mx = -3.4e38f;
  for (int kk = tid; kk < SEQ; kk += 256) mx = fmaxf(mx, sc[kk]);
  red[tid] = mx;
  __syncthreads();
  for (int s = 128; s > 0; s >>= 1) {
    if (tid < s) red[tid] = fmaxf(red[tid], red[tid + s]);
    __syncthreads();
  }
  mx = red[0];
  __syncthreads();

  // exp + sum reduce
  float ls = 0.0f;
  for (int kk = tid; kk < SEQ; kk += 256) {
    float e = expf(sc[kk] - mx);
    sc[kk] = e;
    ls += e;
  }
  float inv = 1.0f / block_sum(ls, red);

  // ctx = probs . v  (each thread: dim d, key chunk of 256)
  const int d = tid & (HDIM - 1);
  const int chunk = tid >> 6;
  float acc = 0.0f;
  for (int kk = chunk * 256; kk < chunk * 256 + 256; ++kk) {
    const float* vp = qkv + ((size_t)(b * SEQ + kk) * QKVD) + h * (3 * HDIM) + 2 * HDIM;
    acc += sc[kk] * vp[d];
  }
  cpart[chunk][d] = acc;
  __syncthreads();
  if (tid < HDIM) {
    float r = (cpart[0][tid] + cpart[1][tid] + cpart[2][tid] + cpart[3][tid]) * inv;
    ctx[((size_t)(b * SEQ + q)) * DMODEL + h * HDIM + tid] = r;
  }
}

// ---------------------------------------------------------------------------
// Host-side orchestration.
// ---------------------------------------------------------------------------
extern "C" void kernel_launch(void* const* d_in, const int* in_sizes, int n_in,
                              void* d_out, int out_size, void* d_ws, size_t ws_size,
                              hipStream_t stream) {
  (void)in_sizes; (void)n_in; (void)out_size; (void)ws_size;

  const int*   ids   = (const int*)d_in[0];
  const float* amask = (const float*)d_in[1];
  const float* wte   = (const float*)d_in[2];
  const float* emb_g = (const float*)d_in[3];
  const float* emb_b = (const float*)d_in[4];
  const float* lnf_g = (const float*)d_in[5];
  const float* lnf_b = (const float*)d_in[6];

  struct Layer {
    const float *ln1g, *ln1b, *wqkv, *bqkv, *wo, *bo, *ln2g, *ln2b, *w1, *b1, *w2, *b2;
  } L[NLAYER];
  for (int l = 0; l < NLAYER; ++l) {
    const int base = 7 + l * 12;
    L[l].ln1g = (const float*)d_in[base + 0];
    L[l].ln1b = (const float*)d_in[base + 1];
    L[l].wqkv = (const float*)d_in[base + 2];
    L[l].bqkv = (const float*)d_in[base + 3];
    L[l].wo   = (const float*)d_in[base + 4];
    L[l].bo   = (const float*)d_in[base + 5];
    L[l].ln2g = (const float*)d_in[base + 6];
    L[l].ln2b = (const float*)d_in[base + 7];
    L[l].w1   = (const float*)d_in[base + 8];
    L[l].b1   = (const float*)d_in[base + 9];
    L[l].w2   = (const float*)d_in[base + 10];
    L[l].b2   = (const float*)d_in[base + 11];
  }

  // Workspace layout
  char* ws = (char*)d_ws;
  size_t off = 0;
  auto alloc = [&](size_t bytes) {
    size_t o = off;
    off += (bytes + 255) & ~(size_t)255;
    return o;
  };
  size_t pw_qkv[NLAYER], pw_o[NLAYER], pw_1[NLAYER], pw_2[NLAYER];
  for (int l = 0; l < NLAYER; ++l) {
    pw_qkv[l] = alloc((size_t)DMODEL * QKVD * 2);
    pw_o[l]   = alloc((size_t)DMODEL * DMODEL * 2);
    pw_1[l]   = alloc((size_t)DMODEL * FFD * 2);
    pw_2[l]   = alloc((size_t)FFD * DMODEL * 2);
  }
  float* hbuf   = (float*)(ws + alloc((size_t)TOKENS * DMODEL * 4));
  float* xbuf   = (float*)(ws + alloc((size_t)TOKENS * DMODEL * 4));
  float* qkvbuf = (float*)(ws + alloc((size_t)TOKENS * QKVD * 4));
  float* ctxbuf = (float*)(ws + alloc((size_t)TOKENS * DMODEL * 4));
  float* attbuf = (float*)(ws + alloc((size_t)TOKENS * DMODEL * 4));
  float* m1buf  = (float*)(ws + alloc((size_t)TOKENS * FFD * 4));
  float* posbuf = (float*)(ws + alloc((size_t)BATCH * SEQ * 4));

  // 1) Pack all weights to bf16 fragment layout (deterministic, every call).
  for (int l = 0; l < NLAYER; ++l) {
    pack_weight_kernel<<<((QKVD / 16) * (DMODEL / 32) * 32) / 256, 256, 0, stream>>>(
        L[l].wqkv, (__bf16*)(ws + pw_qkv[l]), DMODEL, QKVD);
    pack_weight_kernel<<<((DMODEL / 16) * (DMODEL / 32) * 32) / 256, 256, 0, stream>>>(
        L[l].wo, (__bf16*)(ws + pw_o[l]), DMODEL, DMODEL);
    pack_weight_kernel<<<((FFD / 16) * (DMODEL / 32) * 32) / 256, 256, 0, stream>>>(
        L[l].w1, (__bf16*)(ws + pw_1[l]), DMODEL, FFD);
    pack_weight_kernel<<<((DMODEL / 16) * (FFD / 32) * 32) / 256, 256, 0, stream>>>(
        L[l].w2, (__bf16*)(ws + pw_2[l]), FFD, DMODEL);
  }

  // 2) Embedding + LN, ALiBi positions.
  embed_ln_kernel<<<TOKENS, 256, 0, stream>>>(ids, wte, emb_g, emb_b, hbuf);
  positions_kernel<<<BATCH, SEQ, 0, stream>>>(amask, posbuf);

  // 3) Transformer layers. GEMM grid = (M/128) * (N/128) blocks of 256.
  const int MB = TOKENS / 128;  // 16
  for (int l = 0; l < NLAYER; ++l) {
    layernorm_kernel<<<TOKENS, 256, 0, stream>>>(hbuf, L[l].ln1g, L[l].ln1b, xbuf);

    wmma_gemm_kernel<<<MB * (QKVD / 128), 256, 0, stream>>>(
        xbuf, (const __bf16*)(ws + pw_qkv[l]), L[l].bqkv, nullptr,
        qkvbuf, TOKENS, DMODEL, QKVD, 0);

    attention_kernel<<<dim3(SEQ, NHEAD, BATCH), 256, 0, stream>>>(
        qkvbuf, amask, posbuf, ctxbuf);

    wmma_gemm_kernel<<<MB * (DMODEL / 128), 256, 0, stream>>>(
        ctxbuf, (const __bf16*)(ws + pw_o[l]), L[l].bo, hbuf,
        attbuf, TOKENS, DMODEL, DMODEL, 2);

    layernorm_kernel<<<TOKENS, 256, 0, stream>>>(attbuf, L[l].ln2g, L[l].ln2b, xbuf);

    wmma_gemm_kernel<<<MB * (FFD / 128), 256, 0, stream>>>(
        xbuf, (const __bf16*)(ws + pw_1[l]), L[l].b1, nullptr,
        m1buf, TOKENS, DMODEL, FFD, 1);

    wmma_gemm_kernel<<<MB * (DMODEL / 128), 256, 0, stream>>>(
        m1buf, (const __bf16*)(ws + pw_2[l]), L[l].b2, attbuf,
        hbuf, TOKENS, FFD, DMODEL, 2);
  }

  // 4) Final LayerNorm -> output (f32).
  layernorm_kernel<<<TOKENS, 256, 0, stream>>>(hbuf, lnf_g, lnf_b, (float*)d_out);
}
